// HybridSNN_50199577756198
// MI455X (gfx1250) — matrix operand order repaired and checked by reading, not compiled, not run on hardware
//
#include <hip/hip_runtime.h>
#include <hip/hip_bf16.h>

typedef __attribute__((ext_vector_type(16))) _Float16 v16h;
typedef __attribute__((ext_vector_type(8)))  _Float16 v8h;
typedef __attribute__((ext_vector_type(8)))  float    v8f;

// ---------------- sizes ----------------
#define B_      256
#define L0      4097
#define C0      16
#define LC0     2049     // conv0 out
#define LP0     1025     // pool0 out
#define C1      32
#define LC1     513      // conv1 out
#define LP1     257      // pool1 out
#define C2      64
#define LC2     129      // conv2 out
#define TSTEPS  16
#define NPOS    (B_*LC2) // 33024

// workspace layout (bytes)
#define H0_OFF   0u
#define Y_OFF    16793600u                   // 256*16*1025*4
#define S1_OFF   33603584u                   // Y_OFF + 256*32*513*4
#define HS_OFF   37814272u                   // S1_OFF + 256*32*257*2
#define HS_N     (TSTEPS*B_*C2)              // 262144 floats

// ---------------- K0: zero the spike-mean accumulator ----------------
__global__ void k0_zero(float* __restrict__ p, int n) {
    int i = blockIdx.x * 256 + threadIdx.x;
    if (i < n) p[i] = 0.0f;
}

// ---------------- K1: conv0 + BN + ReLU + maxpool (fused) ----------------
__global__ __launch_bounds__(256) void k1_stem(
    const float* __restrict__ x, const float* __restrict__ w0, const float* __restrict__ b0,
    const float* __restrict__ bn_g, const float* __restrict__ bn_b,
    const float* __restrict__ bn_m, const float* __restrict__ bn_v,
    float* __restrict__ h0)
{
    int idx = blockIdx.x * 256 + threadIdx.x;
    if (idx >= B_ * C0 * LP0) return;
    int q = idx % LP0;
    int c = (idx / LP0) % C0;
    int b = idx / (LP0 * C0);
    const float* xb = x + (long)b * L0;

    float w[7];
#pragma unroll
    for (int i = 0; i < 7; ++i) w[i] = w0[c * 7 + i];
    float bias  = b0[c];
    float scale = bn_g[c] * rsqrtf(bn_v[c] + 1e-5f);
    float shift = bn_b[c] - bn_m[c] * scale;

    float best = -INFINITY;
#pragma unroll
    for (int d = -1; d <= 1; ++d) {
        int j = 2 * q + d;                 // conv0 output index, window of pool
        if (j < 0 || j >= LC0) continue;
        float s = bias;
        int base = 2 * j - 3;              // x index of tap 0 (pad 3)
#pragma unroll
        for (int t = 0; t < 7; ++t) {
            int p = base + t;
            if (p >= 0 && p < L0) s = fmaf(xb[p], w[t], s);
        }
        float r = fmaxf(s * scale + shift, 0.0f);   // BN + ReLU
        best = fmaxf(best, r);
    }
    h0[idx] = best;
}

// ---------------- K2: conv1 (16->32, k5, s2, p2) once (T-broadcast collapsed) ----------------
__global__ __launch_bounds__(256) void k2_conv1(
    const float* __restrict__ h0, const float* __restrict__ w1, const float* __restrict__ b1,
    float* __restrict__ y)
{
    __shared__ float Wl[C1 * C0 * 5];           // 2560 floats
    for (int i = threadIdx.x; i < C1 * C0 * 5; i += 256) Wl[i] = w1[i];
    __syncthreads();

    int idx = blockIdx.x * 256 + threadIdx.x;
    if (idx >= B_ * C1 * LC1) return;
    int l  = idx % LC1;
    int oc = (idx / LC1) % C1;
    int b  = idx / (LC1 * C1);

    const float* hb = h0 + (long)b * C0 * LP0;
    const float* wr = &Wl[oc * C0 * 5];
    float s = b1[oc];
    int base = 2 * l - 2;                       // pad 2
#pragma unroll
    for (int ic = 0; ic < C0; ++ic) {
        const float* hc = hb + ic * LP0;
#pragma unroll
        for (int tap = 0; tap < 5; ++tap) {
            int p = base + tap;
            if (p >= 0 && p < LP0) s = fmaf(hc[p], wr[ic * 5 + tap], s);
        }
    }
    y[idx] = s;
}

// ---------------- K3: LIF(16 steps, constant drive) -> bitmask, then maxpool (OR) ----------------
__device__ __forceinline__ unsigned lif16_mask(float x) {
    float v = 0.0f;
    unsigned m = 0;
#pragma unroll
    for (int t = 0; t < TSTEPS; ++t) {
        v = 0.5f * (v + x);                    // v += (x - v)/TAU, TAU = 2
        if (v >= 1.0f) { m |= (1u << t); v = 0.0f; }
    }
    return m;
}

__global__ __launch_bounds__(256) void k3_lif_pool(
    const float* __restrict__ y, unsigned short* __restrict__ s1)
{
    int idx = blockIdx.x * 256 + threadIdx.x;
    if (idx >= B_ * C1 * LP1) return;
    int q = idx % LP1;
    int rest = idx / LP1;                       // b*32 + c
    const float* yc = y + (long)rest * LC1;
    unsigned m = 0;
#pragma unroll
    for (int d = -1; d <= 1; ++d) {
        int j = 2 * q + d;
        if (j >= 0 && j < LC1) m |= lif16_mask(yc[j]);
    }
    s1[idx] = (unsigned short)m;
}

// ---------------- K4: conv2 as WMMA GEMM per t + LIF + length-mean accumulation ----------------
// GEMM: D[oc,n] = W2[oc,k] * S[k,n,t],  oc=64, k=96 (ic*3+tap), n = b*129 + l (33024)
// Per block: 32 positions, all 64 oc, all 16 t. 8 waves = 4 M-frags x 2 N-frags.
//
// Spike bits are converted to f16 with a 2-op trick: bit t of each u16 mask half is
// shifted to bit 10, yielding f16 0x0400 = 2^-14 (or 0). Weights staged into LDS are
// pre-scaled by 2^14 (exact power-of-2 scaling in f16), so the f32 WMMA accumulator
// recovers w*s exactly.
__global__ __launch_bounds__(256) void k4_conv2_wmma(
    const float* __restrict__ w2,               // [64][96] row-major == [oc][ic][tap]
    const float* __restrict__ b2,               // [64]
    const unsigned short* __restrict__ s1,      // [256][32][257] spike bitmasks
    float* __restrict__ hsum)                   // [16][256][64] spike-count accumulator
{
    __shared__ alignas(16) _Float16       Alds[C2 * 96];   // 12 KB, f16 weights * 2^14
    __shared__ alignas(16) unsigned short Mlds[32 * 96];   // 6 KB, im2col'd masks

    int tid = threadIdx.x;
    // stage weights as f16, pre-scaled by 2^14 to compensate spike value 2^-14
    for (int i = tid; i < C2 * 96; i += 256) Alds[i] = (_Float16)(w2[i] * 16384.0f);
    // stage spike masks for this tile of 32 positions (zero-padded at edges)
    int n0 = blockIdx.x * 32;
    for (int i = tid; i < 32 * 96; i += 256) {
        int nn = i / 96, k = i % 96;
        int n = n0 + nn;
        int b = n / LC2, l = n % LC2;
        int ic = k / 3, tap = k % 3;
        int p = 2 * l - 1 + tap;                // pad 1
        unsigned short m = 0;
        if (p >= 0 && p < LP1) m = s1[((long)b * C1 + ic) * LP1 + p];
        Mlds[i] = m;
    }
    __syncthreads();

    int wave = tid >> 5;
    int lane = tid & 31;
    int mi   = wave & 3;        // oc block: rows mi*16 .. mi*16+15
    int ni   = wave >> 2;       // n sub-tile: 0 or 1
    int hf   = lane >> 4;       // hi/lo lane group
    int lm   = lane & 15;

    // per-lane bias for its 8 C-rows (C layout: VGPR r -> M = r + hf*8)
    v8f bias;
#pragma unroll
    for (int r = 0; r < 8; ++r) bias[r] = b2[mi * 16 + r + hf * 8];

    // ---- hoist t-invariant A fragments into registers ----
    // A fragment (16-bit A 16x32): lanes0-15 K 0..15, lanes16-31 K 16..31
    v16h afr[3];
#pragma unroll
    for (int ks = 0; ks < 3; ++ks) {
        int k0 = ks * 32;
        const v8h* pa0 = (const v8h*)&Alds[lm * 96 + k0 + hf * 16];
        const v8h* pa1 = (const v8h*)&Alds[lm * 96 + k0 + hf * 16 + 8];
        v8h alo = *pa0, ahi = *pa1;
#pragma unroll
        for (int i = 0; i < 8; ++i) { afr[ks][i] = alo[i]; afr[ks][i + 8] = ahi[i]; }
    }

    // ---- hoist t-invariant B bitmasks into registers (8 dwords = 16 u16 per k-step) ----
    // B fragment (32x16): column n = lane%16, lanes0-15 K 0..15, lanes16-31 K 16..31
    unsigned msk[3][8];
#pragma unroll
    for (int ks = 0; ks < 3; ++ks) {
        const uint4* pm = (const uint4*)&Mlds[(ni * 16 + lm) * 96 + ks * 32 + hf * 16];
        uint4 m0 = pm[0], m1 = pm[1];
        msk[ks][0] = m0.x; msk[ks][1] = m0.y; msk[ks][2] = m0.z; msk[ks][3] = m0.w;
        msk[ks][4] = m1.x; msk[ks][5] = m1.y; msk[ks][6] = m1.z; msk[ks][7] = m1.w;
    }

    // per-lane output position (C layout: column N = lane % 16)
    int npos = n0 + ni * 16 + lm;
    int bb   = npos / LC2;
    float* hb = hsum + (long)bb * C2;           // + t*B_*C2 + oc

    v8f v = {};                                 // LIF membrane state (8 output elems/lane)

#pragma unroll
    for (int t = 0; t < TSTEPS; ++t) {
        v8f acc = {};
#pragma unroll
        for (int ks = 0; ks < 3; ++ks) {
            // bit t of each u16 half -> bit 10 (f16 0x0400 = 2^-14), packed pair per dword
            union { v16h h; unsigned u[8]; } bf;
#pragma unroll
            for (int i = 0; i < 8; ++i) {
                unsigned m = msk[ks][i];
                bf.u[i] = (t <= 10) ? ((m << (10 - t)) & 0x04000400u)
                                    : ((m >> (t - 10)) & 0x04000400u);
            }
            acc = __builtin_amdgcn_wmma_f32_16x16x32_f16(
                false, afr[ks], false, bf.h, (short)0, acc, false, false);
        }
        // conv2 out + bias -> LIF step, spike -> accumulate length-mean numerator
        v8f xin = acc + bias;
        v = (v + xin) * 0.5f;
#pragma unroll
        for (int r = 0; r < 8; ++r) {
            if (v[r] >= 1.0f) {
                v[r] = 0.0f;
                atomicAdd(&hb[(long)t * B_ * C2 + mi * 16 + r + hf * 8], 1.0f);
            }
        }
    }
}

// ---------------- K5: fc1 + LIF + fc2 + mean over T ----------------
__global__ __launch_bounds__(32) void k5_head(
    const float* __restrict__ hsum, const float* __restrict__ fc1w, const float* __restrict__ fc1b,
    const float* __restrict__ fc2w, const float* __restrict__ fc2b, float* __restrict__ out)
{
    __shared__ float hrow[C2];
    __shared__ float sp[32];
    int b = blockIdx.x;
    int j = threadIdx.x;       // feature 0..31
    float v = 0.0f;
    float accO = 0.0f;
    const float inv129 = 1.0f / 129.0f;

    for (int t = 0; t < TSTEPS; ++t) {
        const float* hr = hsum + ((long)t * B_ + b) * C2;
        hrow[j]      = hr[j]      * inv129;
        hrow[j + 32] = hr[j + 32] * inv129;
        __syncthreads();
        float s = fc1b[j];
#pragma unroll
        for (int d = 0; d < C2; ++d) s = fmaf(fc1w[j * C2 + d], hrow[d], s);
        v = 0.5f * (v + s);
        float spike = (v >= 1.0f) ? 1.0f : 0.0f;
        v *= (1.0f - spike);
        sp[j] = spike;
        __syncthreads();
        if (j < 5) {
            float a = 0.0f;
#pragma unroll
            for (int d = 0; d < 32; ++d) a = fmaf(fc2w[j * 32 + d], sp[d], a);
            accO += a;
        }
        __syncthreads();
    }
    if (j < 5) out[b * 5 + j] = accO * (1.0f / 16.0f) + fc2b[j];
}

// ---------------- launcher ----------------
extern "C" void kernel_launch(void* const* d_in, const int* in_sizes, int n_in,
                              void* d_out, int out_size, void* d_ws, size_t ws_size,
                              hipStream_t stream) {
    const float* x     = (const float*)d_in[0];
    const float* w0    = (const float*)d_in[1];
    const float* b0    = (const float*)d_in[2];
    const float* bn_g  = (const float*)d_in[3];
    const float* bn_b  = (const float*)d_in[4];
    const float* bn_m  = (const float*)d_in[5];
    const float* bn_v  = (const float*)d_in[6];
    const float* w1    = (const float*)d_in[7];
    const float* b1    = (const float*)d_in[8];
    const float* w2    = (const float*)d_in[9];
    const float* b2    = (const float*)d_in[10];
    const float* fc1w  = (const float*)d_in[11];
    const float* fc1b  = (const float*)d_in[12];
    const float* fc2w  = (const float*)d_in[13];
    const float* fc2b  = (const float*)d_in[14];

    char* ws = (char*)d_ws;
    float*          h0   = (float*)(ws + H0_OFF);
    float*          y    = (float*)(ws + Y_OFF);
    unsigned short* s1   = (unsigned short*)(ws + S1_OFF);
    float*          hsum = (float*)(ws + HS_OFF);
    float*          out  = (float*)d_out;

    k0_zero<<<(HS_N + 255) / 256, 256, 0, stream>>>(hsum, HS_N);
    k1_stem<<<(B_ * C0 * LP0 + 255) / 256, 256, 0, stream>>>(x, w0, b0, bn_g, bn_b, bn_m, bn_v, h0);
    k2_conv1<<<(B_ * C1 * LC1 + 255) / 256, 256, 0, stream>>>(h0, w1, b1, y);
    k3_lif_pool<<<(B_ * C1 * LP1 + 255) / 256, 256, 0, stream>>>(y, s1);
    k4_conv2_wmma<<<NPOS / 32, 256, 0, stream>>>(w2, b2, s1, hsum);
    k5_head<<<B_, 32, 0, stream>>>(hsum, fc1w, fc1b, fc2w, fc2b, out);
}
